// GRU_32770600468735
// MI455X (gfx1250) — matrix operand order repaired and checked by reading, not compiled, not run on hardware
//
#include <hip/hip_runtime.h>
#include <hip/hip_bf16.h>
#include <math.h>

typedef __attribute__((ext_vector_type(16))) _Float16 v16h;
typedef __attribute__((ext_vector_type(4)))  _Float16 v4h;
typedef __attribute__((ext_vector_type(8)))  float    v8f;
typedef __attribute__((ext_vector_type(4)))  float    v4f;

#define B_  64
#define S_  2048
#define D_  64
#define H_  128
#define G_  384   // 3*H
#define C_  10

// D = A(16x32 f16) * B(32x16 f16) + C(16x16 f32)
__device__ __forceinline__ v8f wmma_f16(v16h a, v16h b, v8f c) {
  return __builtin_amdgcn_wmma_f32_16x16x32_f16(false, a, false, b, (short)0, c,
                                                false, false);
}

__device__ __forceinline__ float fast_tanh(float x) {
#if __has_builtin(__builtin_amdgcn_tanhf)
  return __builtin_amdgcn_tanhf(x);
#elif __has_builtin(__builtin_amdgcn_tanh_f32)
  return __builtin_amdgcn_tanh_f32(x);
#else
  return tanhf(x);
#endif
}

__device__ __forceinline__ float fast_sigmoid(float x) {
  return __builtin_amdgcn_rcpf(1.0f + __expf(-x));
}

// ---------------------------------------------------------------------------
// Elementwise f32 -> f16 (n4 = #float4 groups; all our sizes divide by 4).
// ---------------------------------------------------------------------------
__global__ __launch_bounds__(256) void cvt_f16_kernel(
    const float* __restrict__ in, _Float16* __restrict__ out, int n4)
{
  const int i = blockIdx.x * 256 + threadIdx.x;
  if (i < n4) {
    const v4f v = ((const v4f*)in)[i];
    v4h o = {(_Float16)v.x, (_Float16)v.y, (_Float16)v.z, (_Float16)v.w};
    ((v4h*)out)[i] = o;
  }
}

// ---------------------------------------------------------------------------
// Input projection GEMM: Y[M,384] = X16[M,K] * W16[384,K]^T + bias (f32 out).
// Grid: (M/128, 384/64), block 256 (8 waves). Wave w: rows [16w,16w+16),
// 4 N-tiles of 16. K in {64,128}. Pure f16 loads -> WMMA, f32 accumulate.
// ---------------------------------------------------------------------------
__global__ __launch_bounds__(256) void proj_gemm_kernel(
    const _Float16* __restrict__ X, const _Float16* __restrict__ W,
    const float* __restrict__ bias, float* __restrict__ Y, int K)
{
  const int lane    = threadIdx.x & 31;
  const int wave    = threadIdx.x >> 5;
  const int halfSel = lane >> 4;      // 0: lanes 0-15, 1: lanes 16-31
  const int l15     = lane & 15;
  const int mbase   = blockIdx.x * 128 + wave * 16;
  const int nbase   = blockIdx.y * 64;

  v8f acc[4] = {};

  const int nkc = K >> 5;
  for (int kc = 0; kc < nkc; ++kc) {
    // A fragment: lane row = l15,
    // halves 0..15 -> K = kc*32 + halfSel*16 + i (contiguous, matches B).
    const _Float16* xrow = X + (size_t)(mbase + l15) * K + kc * 32 + halfSel * 16;
    v16h a;
#pragma unroll
    for (int i = 0; i < 8; ++i) {
      a[i]     = xrow[i];
      a[8 + i] = xrow[8 + i];
    }
#pragma unroll
    for (int nt = 0; nt < 4; ++nt) {
      // B fragment: lane col = nbase+nt*16+l15, 16 contiguous K per half-wave.
      const _Float16* wrow =
          W + (size_t)(nbase + nt * 16 + l15) * K + kc * 32 + halfSel * 16;
      v16h b;
#pragma unroll
      for (int i = 0; i < 16; ++i) b[i] = wrow[i];
      acc[nt] = wmma_f16(a, b, acc[nt]);
    }
  }

  // C/D layout: VGPR v holds row v (lanes 0-15) / row 8+v (lanes 16-31).
#pragma unroll
  for (int nt = 0; nt < 4; ++nt) {
    const int col = nbase + nt * 16 + l15;
    const float bv = bias[col];
#pragma unroll
    for (int v = 0; v < 8; ++v) {
      const int m = mbase + v + halfSel * 8;
      Y[(size_t)m * G_ + col] = acc[nt][v] + bv;
    }
  }
}

// ---------------------------------------------------------------------------
// GRU recurrence over S_ steps for a 16-batch tile. Grid: 4 blocks of 768
// threads (24 waves). Wave w owns output columns [16w, 16w+16): W_hh
// B-fragments pinned in VGPRs; hidden state in LDS (f32 + f16 mirror);
// xg stream register-double-buffered one step ahead.
// ---------------------------------------------------------------------------
__global__ __launch_bounds__(768) void gru_recur_kernel(
    const float* __restrict__ xg,   // [B_*S_, G_], b_ih already folded in
    const float* __restrict__ Whh,  // [G_, H_]
    const float* __restrict__ bhh,  // [G_]
    _Float16* __restrict__ hseq16,  // [B_*S_, H_] f16 or null
    float* __restrict__ hlast)      // [B_, H_] or null
{
  __shared__ _Float16 h16[16 * H_];   // 4 KB, WMMA A source
  __shared__ float    hf [16 * H_];   // 8 KB, exact f32 state
  __shared__ float    hg [16 * G_];   // 24 KB, h @ W_hh^T per step

  const int tid     = threadIdx.x;
  const int lane    = tid & 31;
  const int wave    = tid >> 5;       // 0..23
  const int halfSel = lane >> 4;
  const int l15     = lane & 15;
  const int mb      = blockIdx.x * 16;

  for (int e = tid; e < 16 * H_; e += 768) {
    hf[e]  = 0.0f;
    h16[e] = (_Float16)0.0f;
  }

  // Pin B fragments: B[k][n] = Whh[n][k], one N-tile per wave, K=128.
  v16h Bf[4];
  {
    const float* wr = Whh + (size_t)(wave * 16 + l15) * H_;
#pragma unroll
    for (int kc = 0; kc < 4; ++kc) {
      const int kb = kc * 32 + halfSel * 16;
#pragma unroll
      for (int i = 0; i < 16; ++i) Bf[kc][i] = (_Float16)wr[kb + i];
    }
  }

  // Per-thread gate-element constants (16*128 elems / 768 threads -> 3 its).
  float br[3], bz[3], bn[3];
  const float* xp[3];
  _Float16* sp[3];
  bool valid[3];
#pragma unroll
  for (int it = 0; it < 3; ++it) {
    const int e = tid + it * 768;
    valid[it] = (e < 16 * H_);
    const int m = (e >> 7) & 15;
    const int j = e & (H_ - 1);
    if (valid[it]) { br[it] = bhh[j]; bz[it] = bhh[H_ + j]; bn[it] = bhh[2 * H_ + j]; }
    else           { br[it] = bz[it] = bn[it] = 0.0f; }
    xp[it] = xg + (size_t)(mb + m) * S_ * G_ + j;
    sp[it] = hseq16 ? hseq16 + (size_t)(mb + m) * S_ * H_ + j : (_Float16*)0;
  }

  // Register double-buffer of the xg stream: load step 0 now.
  float xr_c[3], xz_c[3], xn_c[3];
#pragma unroll
  for (int it = 0; it < 3; ++it) {
    if (valid[it]) { xr_c[it] = xp[it][0]; xz_c[it] = xp[it][H_]; xn_c[it] = xp[it][2 * H_]; }
    else           { xr_c[it] = xz_c[it] = xn_c[it] = 0.0f; }
  }

  __syncthreads();

  for (int t = 0; t < S_; ++t) {
    // ---- issue next step's xg loads (consumed after this step's gates) ----
    float xr_n[3], xz_n[3], xn_n[3];
    const bool pf = (t + 1 < S_);
#pragma unroll
    for (int it = 0; it < 3; ++it) {
      if (pf && valid[it]) {
        const float* p = xp[it] + (size_t)(t + 1) * G_;
        xr_n[it] = p[0]; xz_n[it] = p[H_]; xn_n[it] = p[2 * H_];
      } else { xr_n[it] = xz_n[it] = xn_n[it] = 0.0f; }
    }

    // ---- hg = h @ W_hh^T : 4 WMMAs per wave -------------------------------
    v8f acc = {};
#pragma unroll
    for (int kc = 0; kc < 4; ++kc) {
      const int base = l15 * H_ + kc * 32 + halfSel * 16;
      v16h a;
#pragma unroll
      for (int i = 0; i < 8; ++i) {
        a[i]     = h16[base + i];
        a[8 + i] = h16[base + 8 + i];
      }
      acc = wmma_f16(a, Bf[kc], acc);
    }
    {
      const int c0 = wave * 16 + l15;
#pragma unroll
      for (int v = 0; v < 8; ++v)
        hg[(v + halfSel * 8) * G_ + c0] = acc[v];
    }
    __syncthreads();

    // ---- fused GRU gates --------------------------------------------------
#pragma unroll
    for (int it = 0; it < 3; ++it) {
      const int e = tid + it * 768;
      if (e < 16 * H_) {
        const int m = e >> 7;
        const int j = e & (H_ - 1);
        const float hr = hg[m * G_ + j]          + br[it];
        const float hz = hg[m * G_ + H_ + j]     + bz[it];
        const float hn = hg[m * G_ + 2 * H_ + j] + bn[it];
        const float r = fast_sigmoid(xr_c[it] + hr);
        const float z = fast_sigmoid(xz_c[it] + hz);
        const float n = fast_tanh(xn_c[it] + r * hn);
        const float hnew = (1.0f - z) * n + z * hf[e];
        hf[e]  = hnew;
        h16[e] = (_Float16)hnew;
        if (sp[it]) sp[it][(size_t)t * H_] = (_Float16)hnew;
      }
      xr_c[it] = xr_n[it]; xz_c[it] = xz_n[it]; xn_c[it] = xn_n[it];
    }
    __syncthreads();
  }

  if (hlast) {
    for (int e = tid; e < 16 * H_; e += 768)
      hlast[(size_t)(mb + (e >> 7)) * H_ + (e & (H_ - 1))] = hf[e];
  }
}

// ---------------------------------------------------------------------------
// logits = hlast @ fc_w^T + fc_b ; out = log_softmax(logits). 64 blocks x 32.
// ---------------------------------------------------------------------------
__global__ __launch_bounds__(32) void fc_logsoftmax_kernel(
    const float* __restrict__ hlast, const float* __restrict__ fcw,
    const float* __restrict__ fcb, float* __restrict__ out)
{
  __shared__ float logit[C_];
  __shared__ float red[2];
  const int b = blockIdx.x;
  const int j = threadIdx.x;
  if (j < C_) {
    float acc = fcb[j];
    const float* hr = hlast + (size_t)b * H_;
    const float* wr = fcw + (size_t)j * H_;
#pragma unroll 4
    for (int k = 0; k < H_; ++k) acc += hr[k] * wr[k];
    logit[j] = acc;
  }
  __syncthreads();
  if (j == 0) {
    float mx = logit[0];
    for (int i = 1; i < C_; ++i) mx = fmaxf(mx, logit[i]);
    float s = 0.0f;
    for (int i = 0; i < C_; ++i) s += __expf(logit[i] - mx);
    red[0] = mx;
    red[1] = __logf(s);
  }
  __syncthreads();
  if (j < C_) out[(size_t)b * C_ + j] = logit[j] - red[0] - red[1];
}

// ---------------------------------------------------------------------------
extern "C" void kernel_launch(void* const* d_in, const int* in_sizes, int n_in,
                              void* d_out, int out_size, void* d_ws,
                              size_t ws_size, hipStream_t stream) {
  (void)in_sizes; (void)n_in; (void)out_size; (void)ws_size;
  const float* x     = (const float*)d_in[0];
  const float* W_ih0 = (const float*)d_in[1];
  const float* W_hh0 = (const float*)d_in[2];
  const float* b_ih0 = (const float*)d_in[3];
  const float* b_hh0 = (const float*)d_in[4];
  const float* W_ih1 = (const float*)d_in[5];
  const float* W_hh1 = (const float*)d_in[6];
  const float* b_ih1 = (const float*)d_in[7];
  const float* b_hh1 = (const float*)d_in[8];
  const float* fc_w  = (const float*)d_in[9];
  const float* fc_b  = (const float*)d_in[10];

  const int M = B_ * S_;

  char* ws = (char*)d_ws;
  float*    xg    = (float*)ws;     ws += (size_t)M * G_ * sizeof(float);
  _Float16* x16   = (_Float16*)ws;  ws += (size_t)M * D_ * sizeof(_Float16);
  _Float16* h116  = (_Float16*)ws;  ws += (size_t)M * H_ * sizeof(_Float16);
  _Float16* wih0h = (_Float16*)ws;  ws += (size_t)G_ * D_ * sizeof(_Float16);
  _Float16* wih1h = (_Float16*)ws;  ws += (size_t)G_ * H_ * sizeof(_Float16);
  float*    hlast = (float*)ws;

  // Pre-convert f32 -> f16 for all WMMA A/B inputs of the projection GEMMs.
  cvt_f16_kernel<<<(M * D_ / 4 + 255) / 256, 256, 0, stream>>>(x, x16, M * D_ / 4);
  cvt_f16_kernel<<<(G_ * D_ / 4 + 255) / 256, 256, 0, stream>>>(W_ih0, wih0h, G_ * D_ / 4);
  cvt_f16_kernel<<<(G_ * H_ / 4 + 255) / 256, 256, 0, stream>>>(W_ih1, wih1h, G_ * H_ / 4);

  dim3 pgrid(M / 128, G_ / 64);
  proj_gemm_kernel<<<pgrid, 256, 0, stream>>>(x16, wih0h, b_ih0, xg, D_);
  gru_recur_kernel<<<4, 768, 0, stream>>>(xg, W_hh0, b_hh0, h116, nullptr);
  proj_gemm_kernel<<<pgrid, 256, 0, stream>>>(h116, wih1h, b_ih1, xg, H_);
  gru_recur_kernel<<<4, 768, 0, stream>>>(xg, W_hh1, b_hh1, nullptr, hlast);
  fc_logsoftmax_kernel<<<B_, 32, 0, stream>>>(hlast, fc_w, fc_b,
                                              (float*)d_out);
}
